// Attention_33672543601334
// MI455X (gfx1250) — matrix operand order repaired and checked by reading, not compiled, not run on hardware
//
#include <hip/hip_runtime.h>
#include <hip/hip_bf16.h>

// ---------------- CDNA5 (gfx1250) fused multi-head attention ----------------
// 1) qkv_proj_kernel   : Q/K/V = x @ W^T      f32 -> bf16 WMMA (f32 acc) -> bf16 [b,h,n,dh]
//                        (Q is pre-scaled by 512^-0.5 so flash S = QK^T + mask directly)
// 2) flash_attn_kernel : softmax(Q K^T + mask) V, streaming (online softmax);
//                        mask enters through the WMMA C operand. K-tile and
//                        mask-tile staged by the Tensor Data Mover (TDM) with
//                        hardware LDS row padding; V^T via lane scatter.
// 3) out_proj_kernel   : out = att @ Wo^T + bo, bf16 WMMA -> f32
//
// bf16 operands / f32 accumulation (V_WMMA_F32_16X16X32_BF16): 8x the
// FLOPs/instr of V_WMMA_F32_16X16X4_F32; 512-deep dots keep precision with
// f32 accumulate. Conversions use native __bf16 casts (v_cvt_pk_bf16_f32 /
// v_fma_mix*_bf16). Pure-copy staging uses GLOBAL_LOAD_ASYNC_TO_LDS_B128.

#define OBS_D   512
#define HEADS   8
#define SEQLEN  2048
#define BATCH   2
#define DHEAD   64
#define ROWS    (BATCH*SEQLEN)          // 4096
#define SCALE_F 0.04419417382415922f    // 512^-0.5 (module scales by obs_dim^-0.5)

typedef __attribute__((ext_vector_type(16))) __bf16          bf16x16;
typedef __attribute__((ext_vector_type(8)))  float           f32x8;
typedef __attribute__((ext_vector_type(8)))  unsigned short  u16x8;
typedef __attribute__((ext_vector_type(16))) unsigned short  u16x16;
typedef __attribute__((__vector_size__(4 * sizeof(int)))) int i32x4;
typedef __attribute__((ext_vector_type(4)))  unsigned int    u32x4;
typedef __attribute__((ext_vector_type(8)))  int             i32x8;

union FragBF { u16x16 u; bf16x16 b; };

// Native f32 -> bf16 (RNE) via hardware convert, reinterpreted as raw bits.
__device__ __forceinline__ unsigned short f32_to_bf16(float f) {
  return __builtin_bit_cast(unsigned short, (__bf16)f);
}

__device__ __forceinline__ bf16x16 make_frag(u16x8 lo, u16x8 hi) {
  FragBF fr;
#pragma unroll
  for (int e = 0; e < 8; ++e) { fr.u[e] = lo[e]; fr.u[8 + e] = hi[e]; }
  return fr.b;
}

__device__ __forceinline__ f32x8 wmma_bf16(bf16x16 a, bf16x16 b, f32x8 c) {
  // v_wmma_f32_16x16x32_bf16  (neg_a, A, neg_b, B, c_mod, C, reuse_a, reuse_b)
  return __builtin_amdgcn_wmma_f32_16x16x32_bf16(false, a, false, b, (short)0, c,
                                                 false, false);
}

// A-fragment (16x32, bf16): lane holds row (lane&15); low lanes K {0..7,16..23},
// high lanes K {8..15,24..31}  -> base offset (hi?8:0), chunks at +0 and +16.
__device__ __forceinline__ bf16x16 load_fragA(const unsigned short* p, int lhi) {
  const unsigned short* ap = p + (lhi ? 8 : 0);
  u16x8 lo = *(const u16x8*)(ap);
  u16x8 hi = *(const u16x8*)(ap + 16);
  return make_frag(lo, hi);
}
// B-fragment (32x16, bf16): lane holds column (lane&15); low lanes K {0..7,16..23},
// high lanes K {8..15,24..31} -> base offset (hi?8:0), chunks +0 and +16.
__device__ __forceinline__ bf16x16 load_fragB(const unsigned short* p, int lhi) {
  const unsigned short* bp = p + (lhi ? 8 : 0);
  u16x8 lo = *(const u16x8*)(bp);
  u16x8 hi = *(const u16x8*)(bp + 16);
  return make_frag(lo, hi);
}

// -------- gfx1250 async global->LDS staging (ASYNCcnt-tracked DMA path) -----
#if __has_builtin(__builtin_amdgcn_global_load_async_to_lds_b128)
#define HAVE_ASYNC_LDS 1
typedef __attribute__((address_space(1))) i32x4* gptr_b128;
typedef __attribute__((address_space(3))) i32x4* lptr_b128;
#else
#define HAVE_ASYNC_LDS 0
#endif

__device__ __forceinline__ void copy16_to_lds(void* lds_dst, const void* gsrc) {
#if HAVE_ASYNC_LDS
  __builtin_amdgcn_global_load_async_to_lds_b128((gptr_b128)gsrc, (lptr_b128)lds_dst,
                                                 0, 0);
#else
  *(uint4*)lds_dst = *(const uint4*)gsrc;
#endif
}
__device__ __forceinline__ void staging_join() {
#if HAVE_ASYNC_LDS
  asm volatile("s_wait_asynccnt 0x0" ::: "memory");
#endif
}

// -------- gfx1250 Tensor Data Mover (TENSORcnt-tracked tile DMA) ------------
#if __has_builtin(__builtin_amdgcn_tensor_load_to_lds) && \
    __has_builtin(__builtin_amdgcn_s_wait_tensorcnt)
#define HAVE_TDM 1
#else
#define HAVE_TDM 0
#endif

#if HAVE_TDM
// LDS byte offset of a __shared__ object (addrspace(3) pointer value).
__device__ __forceinline__ unsigned lds_offset(const void* p) {
  return (unsigned)(unsigned long long)
      (__attribute__((address_space(3))) const void*)p;
}

// Issue a 2D TDM tile load: tile_d1 rows x tile_d0 elements, element size
// (1<<ds_log2) bytes, source row stride `stride0` elements, with hardware LDS
// row padding (pad codes per D# spec: interval 2<<c DWORDs, amount c+1 DWORDs).
__device__ __forceinline__ void tdm_load_2d(
    unsigned lds_addr, const void* gaddr, unsigned ds_log2,
    unsigned tensor_d0, unsigned tensor_d1, unsigned stride0,
    unsigned tile_d0, unsigned tile_d1,
    unsigned pad_interval_code, unsigned pad_amount_code) {
  const unsigned long long ga = (unsigned long long)gaddr;
  u32x4 g0;
  g0[0] = 0x1u;                                      // count=1, user descriptor
  g0[1] = lds_addr;                                  // D#.lds_addr
  g0[2] = (unsigned)(ga & 0xFFFFFFFFu);              // global_addr[31:0]
  g0[3] = (unsigned)((ga >> 32) & 0x01FFFFFFu) | (2u << 30);  // addr[56:32] | type=2
  i32x8 g1;
  g1[0] = (int)((ds_log2 << 16) | (1u << 20) |       // data_size | pad_enable
                (pad_interval_code << 22) | (pad_amount_code << 25));
  g1[1] = (int)((tensor_d0 & 0xFFFFu) << 16);        // tensor_dim0[15:0] @ bits 63:48
  g1[2] = (int)(((tensor_d0 >> 16) & 0xFFFFu) | ((tensor_d1 & 0xFFFFu) << 16));
  g1[3] = (int)(((tensor_d1 >> 16) & 0xFFFFu) | ((tile_d0 & 0xFFFFu) << 16));
  g1[4] = (int)(tile_d1 & 0xFFFFu);                  // tile_dim1 | tile_dim2=0
  g1[5] = (int)stride0;                              // tensor_dim0_stride[31:0]
  g1[6] = 0;                                         // stride0[47:32] | stride1[15:0]
  g1[7] = 0;
  const i32x4 z4 = {0, 0, 0, 0};                     // 2D: groups 2/3 unused
  const i32x8 z8 = {0, 0, 0, 0, 0, 0, 0, 0};         // 6-arg form: extra group
  __builtin_amdgcn_tensor_load_to_lds(g0, g1, z4, z4, z8, 0);
}
#endif

// ======================= 1) QKV projection ================================
// y[r,c] = outScale * sum_k x[r,k] * W[c,k]  (x @ W^T); both operands are
// K-contiguous. Block tile 128x64, 8 waves -> 32x32 per wave, K stepped by 32.
// Output written bf16 into [b, h, n, dh] layout.
__global__ __launch_bounds__(256) void qkv_proj_kernel(
    const float* __restrict__ X, const float* __restrict__ W,
    unsigned short* __restrict__ Out, float outScale) {
  constexpr int TM = 128, TN = 64, TK = 32;
  constexpr int LDA = 40, LDB = 40;                 // bf16 elems; 80B rows (16B-aligned chunks)
  __shared__ unsigned short As[TM * LDA];
  __shared__ unsigned short Bs[TN * LDB];

  const int ntiles_n = OBS_D / TN;                  // 8
  const int mtile = (blockIdx.x / ntiles_n) * TM;
  const int ntile = (blockIdx.x % ntiles_n) * TN;
  const int tid  = threadIdx.x;
  const int lane = tid & 31, wave = tid >> 5;
  const int wm = wave & 3, wn = wave >> 2;          // 4x2 wave grid
  const int llo = lane & 15, lhi = lane >> 4;

  f32x8 acc[2][2] = {};

  for (int k0 = 0; k0 < OBS_D; k0 += TK) {
    { // stage A: 128x32 f32 -> bf16 LDS (16 elems / thread, native cvt)
      const int row = tid >> 1, col = (tid & 1) * 16;
      const float* src = X + (size_t)(mtile + row) * OBS_D + k0 + col;
      unsigned short* dst = As + row * LDA + col;
#pragma unroll
      for (int e = 0; e < 16; e += 4) {
        float4 f = *(const float4*)(src + e);
        dst[e] = f32_to_bf16(f.x); dst[e+1] = f32_to_bf16(f.y);
        dst[e+2] = f32_to_bf16(f.z); dst[e+3] = f32_to_bf16(f.w);
      }
    }
    { // stage B: 64x32 f32 -> bf16 LDS (8 elems / thread)
      const int row = tid >> 2, col = (tid & 3) * 8;
      const float* src = W + (size_t)(ntile + row) * OBS_D + k0 + col;
      unsigned short* dst = Bs + row * LDB + col;
#pragma unroll
      for (int e = 0; e < 8; e += 4) {
        float4 f = *(const float4*)(src + e);
        dst[e] = f32_to_bf16(f.x); dst[e+1] = f32_to_bf16(f.y);
        dst[e+2] = f32_to_bf16(f.z); dst[e+3] = f32_to_bf16(f.w);
      }
    }
    __syncthreads();

    bf16x16 afr[2], bfr[2];
#pragma unroll
    for (int mi = 0; mi < 2; ++mi)
      afr[mi] = load_fragA(As + (wm * 32 + mi * 16 + llo) * LDA, lhi);
#pragma unroll
    for (int ni = 0; ni < 2; ++ni)
      bfr[ni] = load_fragB(Bs + (wn * 32 + ni * 16 + llo) * LDB, lhi);
#pragma unroll
    for (int mi = 0; mi < 2; ++mi)
#pragma unroll
      for (int ni = 0; ni < 2; ++ni)
        acc[mi][ni] = wmma_bf16(afr[mi], bfr[ni], acc[mi][ni]);
    __syncthreads();
  }

  // epilogue: C layout -> bf16 scatter into [b,h,n,dh]
#pragma unroll
  for (int mi = 0; mi < 2; ++mi)
#pragma unroll
    for (int ni = 0; ni < 2; ++ni)
#pragma unroll
      for (int i = 0; i < 8; ++i) {
        const int r = mtile + wm * 32 + mi * 16 + lhi * 8 + i;
        const int c = ntile + wn * 32 + ni * 16 + llo;
        const size_t idx =
            (((size_t)(r >> 11) * HEADS + (c >> 6)) * SEQLEN + (r & (SEQLEN - 1))) * DHEAD +
            (c & (DHEAD - 1));
        Out[idx] = f32_to_bf16(acc[mi][ni][i] * outScale);
      }
}

// ======================= 2) flash attention ================================
// Block = (b, h, 64 query rows); 4 waves x 16 rows each (softmax rows never
// cross a wave). K staged natural-major (B-operand of QK^T), V transposed
// (B-operand of P.V). S accumulator is seeded with the additive mask tile, so
// S = Q K^T + mask comes straight out of the WMMA (Q carries the 512^-0.5).
__global__ __launch_bounds__(128) void flash_attn_kernel(
    const unsigned short* __restrict__ Q, const unsigned short* __restrict__ K,
    const unsigned short* __restrict__ V, const float* __restrict__ Mask,
    unsigned short* __restrict__ AttOut) {
  constexpr int QT = 64, KC = 64;
  constexpr int LK = 72, LV = 72, LP = 72, LM = 68;
  __shared__ unsigned short Ks[KC * LK];       // [key][dh]   (+4-DWORD row pad)
  __shared__ unsigned short Vts[DHEAD * LV];   // [dh][key]   (transposed V)
  __shared__ unsigned short Ps[4 * 16 * LP];   // per-wave P (C->A layout bounce)
  __shared__ float          Ms[QT * LM];       // mask tile f32 (+4-DWORD row pad)

  const int tiles = SEQLEN / QT;               // 32
  const int bh = blockIdx.x / tiles, qt = blockIdx.x % tiles;
  const int tid = threadIdx.x, lane = tid & 31, wave = tid >> 5;
  const int llo = lane & 15, lhi = lane >> 4;

  const size_t bhoff = (size_t)bh * SEQLEN * DHEAD;
  const unsigned short* Qbh = Q + bhoff;
  const unsigned short* Kbh = K + bhoff;
  const unsigned short* Vbh = V + bhoff;
  const int b = bh >> 3, h = bh & 7;
  const int qbase = qt * QT;

  // Q fragments for this wave's 16 rows (K-dim = dh = 64 -> two 16x32 frags)
  bf16x16 qa[2];
  {
    const unsigned short* qp = Qbh + (size_t)(qbase + wave * 16 + llo) * DHEAD;
#pragma unroll
    for (int kk = 0; kk < 2; ++kk) qa[kk] = load_fragA(qp + kk * 32, lhi);
  }

  float m_run[8], lsum[8];
#pragma unroll
  for (int i = 0; i < 8; ++i) { m_run[i] = -3.0e38f; lsum[i] = 0.f; }
  f32x8 o[4] = {};

  for (int kb = 0; kb < SEQLEN; kb += KC) {
    __syncthreads();                                   // compute of prev tile done
#if HAVE_TDM
    if (wave == 0) {
      // K tile: 64 rows x 64 bf16 (128B rows) -> LDS rows padded to 144B
      // (pad_interval code 4 = 32 DWORDs, pad_amount code 3 = 4 DWORDs).
      tdm_load_2d(lds_offset(Ks), Kbh + (size_t)kb * DHEAD, /*ds=*/1,
                  /*t_d0=*/DHEAD, /*t_d1=*/KC, /*stride0=*/DHEAD,
                  /*tile_d0=*/DHEAD, /*tile_d1=*/KC, 4, 3);
      // mask tile: 64 rows x 64 f32 (256B rows) -> LDS rows padded to 272B
      // (pad_interval code 5 = 64 DWORDs, pad_amount code 3 = 4 DWORDs).
      tdm_load_2d(lds_offset(Ms), Mask + (size_t)qbase * SEQLEN + kb, /*ds=*/2,
                  /*t_d0=*/SEQLEN, /*t_d1=*/QT, /*stride0=*/SEQLEN,
                  /*tile_d0=*/QT, /*tile_d1=*/QT, 5, 3);
    }
#else
    { // stage K tile: 64x64 bf16, 32 elems / thread, 16B async copies
      const int row = tid >> 1, half = (tid & 1) * 32;
      const unsigned short* src = Kbh + (size_t)(kb + row) * DHEAD + half;
      unsigned short* dst = Ks + row * LK + half;
#pragma unroll
      for (int e = 0; e < 4; ++e) copy16_to_lds(dst + e * 8, src + e * 8);
    }
    { // stage mask tile 64x64 f32 (additive), 32 f32 / thread
      const int row = tid >> 1, half = (tid & 1) * 32;
      const float* src = Mask + (size_t)(qbase + row) * SEQLEN + kb + half;
      float* dst = Ms + row * LM + half;
#pragma unroll
      for (int e = 0; e < 32; e += 4) copy16_to_lds(dst + e, src + e);
    }
#endif
    if (tid < 64) { // stage V^T: one row per thread, scatter as a column
      const uint4* src = (const uint4*)(Vbh + (size_t)(kb + tid) * DHEAD);
      if (kb + KC < SEQLEN)
        __builtin_prefetch(Vbh + (size_t)(kb + KC + tid) * DHEAD, 0, 1);
#pragma unroll
      for (int e = 0; e < 8; ++e) {
        uint4 v = src[e];
        const int d = e * 8;
        Vts[(d + 0) * LV + tid] = (unsigned short)(v.x);
        Vts[(d + 1) * LV + tid] = (unsigned short)(v.x >> 16);
        Vts[(d + 2) * LV + tid] = (unsigned short)(v.y);
        Vts[(d + 3) * LV + tid] = (unsigned short)(v.y >> 16);
        Vts[(d + 4) * LV + tid] = (unsigned short)(v.z);
        Vts[(d + 5) * LV + tid] = (unsigned short)(v.z >> 16);
        Vts[(d + 6) * LV + tid] = (unsigned short)(v.w);
        Vts[(d + 7) * LV + tid] = (unsigned short)(v.w >> 16);
      }
    }
#if HAVE_TDM
    if (wave == 0) __builtin_amdgcn_s_wait_tensorcnt(0);  // TDM tiles landed
#else
    staging_join();                                       // async copies landed
#endif
    __syncthreads();

    // S = Q K^T + mask : seed the C operand with the mask tile (C layout:
    // element i of sub-tile j is row lhi*8+i, col j*16+llo).
    f32x8 s[4];
#pragma unroll
    for (int j = 0; j < 4; ++j)
#pragma unroll
      for (int i = 0; i < 8; ++i)
        s[j][i] = Ms[(wave * 16 + lhi * 8 + i) * LM + j * 16 + llo];
#pragma unroll
    for (int j = 0; j < 4; ++j)
#pragma unroll
      for (int kk = 0; kk < 2; ++kk)
        s[j] = wmma_bf16(qa[kk], load_fragB(Ks + (j * 16 + llo) * LK + kk * 32, lhi), s[j]);

    // online softmax
    float cmax[8];
#pragma unroll
    for (int i = 0; i < 8; ++i) {
      float v = fmaxf(fmaxf(s[0][i], s[1][i]), fmaxf(s[2][i], s[3][i]));
      v = fmaxf(v, __shfl_xor(v, 1, 32));              // row max across 16-lane group
      v = fmaxf(v, __shfl_xor(v, 2, 32));
      v = fmaxf(v, __shfl_xor(v, 4, 32));
      v = fmaxf(v, __shfl_xor(v, 8, 32));
      cmax[i] = v;
    }
#pragma unroll
    for (int i = 0; i < 8; ++i) {
      const float m_new = fmaxf(m_run[i], cmax[i]);
      const float cf = __expf(m_run[i] - m_new);       // exp(-huge) -> 0 on first tile
      m_run[i] = m_new;
      float psum = 0.f;
#pragma unroll
      for (int j = 0; j < 4; ++j) {
        const float p = __expf(s[j][i] - m_new);
        psum += p;
        Ps[(wave * 16 + lhi * 8 + i) * LP + j * 16 + llo] = f32_to_bf16(p);
      }
      lsum[i] = lsum[i] * cf + psum;
#pragma unroll
      for (int j = 0; j < 4; ++j) o[j][i] *= cf;       // rescale running O
    }
    asm volatile("s_wait_dscnt 0x0" ::: "memory");     // own P writes visible to own reads

    // O += P V  (4 dh sub-tiles x 2 key K-steps)
    const unsigned short* pbase = Ps + wave * 16 * LP;
#pragma unroll
    for (int kk = 0; kk < 2; ++kk) {
      bf16x16 pf = load_fragA(pbase + llo * LP + kk * 32, lhi);
#pragma unroll
      for (int j = 0; j < 4; ++j)
        o[j] = wmma_bf16(pf, load_fragB(Vts + (j * 16 + llo) * LV + kk * 32, lhi), o[j]);
    }
  }

  // finalize: total row sums across the 16-lane group, then O / l
#pragma unroll
  for (int i = 0; i < 8; ++i) {
    float v = lsum[i];
    v += __shfl_xor(v, 1, 32);
    v += __shfl_xor(v, 2, 32);
    v += __shfl_xor(v, 4, 32);
    v += __shfl_xor(v, 8, 32);
    lsum[i] = 1.0f / v;
  }
#pragma unroll
  for (int j = 0; j < 4; ++j)
#pragma unroll
    for (int i = 0; i < 8; ++i) {
      const int qrow = qbase + wave * 16 + lhi * 8 + i;
      const int dcol = h * DHEAD + j * 16 + llo;
      AttOut[((size_t)b * SEQLEN + qrow) * OBS_D + dcol] = f32_to_bf16(o[j][i] * lsum[i]);
    }
}

// ======================= 3) output projection ==============================
__global__ __launch_bounds__(256) void out_proj_kernel(
    const unsigned short* __restrict__ A, const float* __restrict__ W,
    const float* __restrict__ bias, float* __restrict__ Out) {
  constexpr int TM = 128, TN = 64, TK = 32;
  constexpr int LDA = 40, LDB = 40;
  __shared__ unsigned short As[TM * LDA];
  __shared__ unsigned short Bs[TN * LDB];

  const int ntiles_n = OBS_D / TN;
  const int mtile = (blockIdx.x / ntiles_n) * TM;
  const int ntile = (blockIdx.x % ntiles_n) * TN;
  const int tid  = threadIdx.x;
  const int lane = tid & 31, wave = tid >> 5;
  const int wm = wave & 3, wn = wave >> 2;
  const int llo = lane & 15, lhi = lane >> 4;

  f32x8 acc[2][2] = {};

  for (int k0 = 0; k0 < OBS_D; k0 += TK) {
    { // stage A (already bf16): pure copy -> async path
      const int row = tid >> 1, col = (tid & 1) * 16;
      const unsigned short* src = A + (size_t)(mtile + row) * OBS_D + k0 + col;
      unsigned short* dst = As + row * LDA + col;
      copy16_to_lds(dst, src);
      copy16_to_lds(dst + 8, src + 8);
    }
    { // stage B (Wo f32 -> bf16, native cvt)
      const int row = tid >> 2, col = (tid & 3) * 8;
      const float* src = W + (size_t)(ntile + row) * OBS_D + k0 + col;
      unsigned short* dst = Bs + row * LDB + col;
#pragma unroll
      for (int e = 0; e < 8; e += 4) {
        float4 f = *(const float4*)(src + e);
        dst[e] = f32_to_bf16(f.x); dst[e+1] = f32_to_bf16(f.y);
        dst[e+2] = f32_to_bf16(f.z); dst[e+3] = f32_to_bf16(f.w);
      }
    }
    staging_join();
    __syncthreads();

    bf16x16 afr[2], bfr[2];
#pragma unroll
    for (int mi = 0; mi < 2; ++mi)
      afr[mi] = load_fragA(As + (wm * 32 + mi * 16 + llo) * LDA, lhi);
#pragma unroll
    for (int ni = 0; ni < 2; ++ni)
      bfr[ni] = load_fragB(Bs + (wn * 32 + ni * 16 + llo) * LDB, lhi);
#pragma unroll
    for (int mi = 0; mi < 2; ++mi)
#pragma unroll
      for (int ni = 0; ni < 2; ++ni)
        acc[mi][ni] = wmma_bf16(afr[mi], bfr[ni], acc[mi][ni]);
    __syncthreads();
  }

#pragma unroll
  for (int mi = 0; mi < 2; ++mi)
#pragma unroll
    for (int ni = 0; ni < 2; ++ni)
#pragma unroll
      for (int i = 0; i < 8; ++i) {
        const int r = mtile + wm * 32 + mi * 16 + lhi * 8 + i;
        const int c = ntile + wn * 32 + ni * 16 + llo;
        Out[(size_t)r * OBS_D + c] = acc[mi][ni][i] + bias[c];
      }
}

// ======================= host-side launch ==================================
extern "C" void kernel_launch(void* const* d_in, const int* in_sizes, int n_in,
                              void* d_out, int out_size, void* d_ws, size_t ws_size,
                              hipStream_t stream) {
  const float* x    = (const float*)d_in[0];
  const float* mask = (const float*)d_in[1];
  const float* Wq   = (const float*)d_in[2];
  const float* Wk   = (const float*)d_in[3];
  const float* Wv   = (const float*)d_in[4];
  const float* Wo   = (const float*)d_in[5];
  const float* bo   = (const float*)d_in[6];
  float* out = (float*)d_out;

  // workspace: 4 bf16 buffers of 2*8*2048*64 = 2,097,152 elements (4 MiB) each
  const size_t NELT = (size_t)BATCH * HEADS * SEQLEN * DHEAD;
  unsigned short* ws = (unsigned short*)d_ws;
  unsigned short* Qb = ws;
  unsigned short* Kb = ws + NELT;
  unsigned short* Vb = ws + 2 * NELT;
  unsigned short* AO = ws + 3 * NELT;   // [b, n, d] bf16

  const int proj_blocks = (ROWS / 128) * (OBS_D / 64);          // 256
  qkv_proj_kernel<<<proj_blocks, 256, 0, stream>>>(x, Wq, Qb, SCALE_F); // Q pre-scaled
  qkv_proj_kernel<<<proj_blocks, 256, 0, stream>>>(x, Wk, Kb, 1.0f);
  qkv_proj_kernel<<<proj_blocks, 256, 0, stream>>>(x, Wv, Vb, 1.0f);

  const int fa_blocks = BATCH * HEADS * (SEQLEN / 64);          // 512
  flash_attn_kernel<<<fa_blocks, 128, 0, stream>>>(Qb, Kb, Vb, mask, AO);

  out_proj_kernel<<<proj_blocks, 256, 0, stream>>>(AO, Wo, bo, out);
}
